// mBARTFull_18915035972258
// MI455X (gfx1250) — matrix operand order repaired and checked by reading, not compiled, not run on hardware
//
#include <hip/hip_runtime.h>
#include <hip/hip_bf16.h>
#include <math.h>

typedef __bf16 bf16_t;
typedef __attribute__((ext_vector_type(16))) __bf16 v16bf;
typedef __attribute__((ext_vector_type(8)))  __bf16 v8bf;
typedef __attribute__((ext_vector_type(8)))  float  v8f;
typedef __attribute__((ext_vector_type(4)))  float  v4f;

static __device__ __forceinline__ v8f wmma_bf16(v16bf a, v16bf b, v8f c) {
  return __builtin_amdgcn_wmma_f32_16x16x32_bf16(false, a, false, b, (short)0, c,
                                                 false, false);
}

static __device__ __forceinline__ v16bf cat8(v8bf lo, v8bf hi) {
  return __builtin_shufflevector(lo, hi, 0, 1, 2, 3, 4, 5, 6, 7,
                                 8, 9, 10, 11, 12, 13, 14, 15);
}

// ---------------------------------------------------------------- fp32->bf16
__global__ __launch_bounds__(256)
void cvt_f32_bf16_kernel(const float* __restrict__ in, bf16_t* __restrict__ out,
                         int n) {
  int i = (blockIdx.x * 256 + threadIdx.x) * 4;
  if (i < n) {
    v4f v = *(const v4f*)(in + i);
    out[i + 0] = (bf16_t)v.x;
    out[i + 1] = (bf16_t)v.y;
    out[i + 2] = (bf16_t)v.z;
    out[i + 3] = (bf16_t)v.w;
  }
}

// ---------------------------------------------------------------- layernorm
__global__ __launch_bounds__(256)
void layernorm_bf16_kernel(const float* __restrict__ x, const float* __restrict__ g,
                           const float* __restrict__ bta, bf16_t* __restrict__ out) {
  const int E = 1024;
  int row = blockIdx.x;
  const float* xr = x + (size_t)row * E;
  float s = 0.f, s2 = 0.f;
  for (int c = threadIdx.x; c < E; c += 256) {
    float v = xr[c];
    s += v;
    s2 += v * v;
  }
  __shared__ float r0[256], r1[256];
  r0[threadIdx.x] = s;
  r1[threadIdx.x] = s2;
  __syncthreads();
  for (int off = 128; off > 0; off >>= 1) {
    if ((int)threadIdx.x < off) {
      r0[threadIdx.x] += r0[threadIdx.x + off];
      r1[threadIdx.x] += r1[threadIdx.x + off];
    }
    __syncthreads();
  }
  float mean = r0[0] * (1.0f / E);
  float var = r1[0] * (1.0f / E) - mean * mean;
  float rs = rsqrtf(var + 1e-5f);
  for (int c = threadIdx.x; c < E; c += 256)
    out[(size_t)row * E + c] = (bf16_t)((xr[c] - mean) * rs * g[c] + bta[c]);
}

// ---------------------------------------------------------------- gelu (erf)
__global__ __launch_bounds__(256)
void gelu_f32_bf16_kernel(const float* __restrict__ in, bf16_t* __restrict__ out,
                          int n) {
  int i = blockIdx.x * 256 + threadIdx.x;
  if (i < n) {
    float v = in[i];
    out[i] = (bf16_t)(0.5f * v * (1.0f + erff(v * 0.70710678118654752f)));
  }
}

// ---------------------------------------------------------------- WMMA GEMM
// C[M,N] = A[M,K](bf16) * W[N,K](bf16)^T + bias[N] (+ resid[M,N])
// Block: 256 thr = 8 waves (2x4); wave tile 64x64 = 4x4 WMMA tiles.
__global__ __launch_bounds__(256)
void gemm_bf16_wmma_kernel(const bf16_t* __restrict__ A,
                           const bf16_t* __restrict__ W,
                           const float* __restrict__ bias,
                           const float* __restrict__ resid,
                           float* __restrict__ C, int M, int N, int K) {
  const int lane = threadIdx.x & 31;
  const int wave = threadIdx.x >> 5;
  const int lh = lane >> 4;  // half-wave id
  const int ll = lane & 15;
  const int mBase = blockIdx.y * 128 + (wave >> 2) * 64;
  const int nBase = blockIdx.x * 256 + (wave & 3) * 64;

  v8f acc[4][4] = {};
  const bf16_t* Ar[4];
  const bf16_t* Wr[4];
#pragma unroll
  for (int t = 0; t < 4; ++t) {
    Ar[t] = A + (size_t)(mBase + t * 16 + ll) * K;
    Wr[t] = W + (size_t)(nBase + t * 16 + ll) * K;
  }

  for (int k = 0; k < K; k += 32) {
    v16bf a[4], b[4];
#pragma unroll
    for (int t = 0; t < 4; ++t) {
      // A fragment 16x32: lanes 0-15 rows m, K={0..7,16..23}; lanes 16-31 K=+8
      v8bf lo = *(const v8bf*)(Ar[t] + k + 8 * lh);
      v8bf hi = *(const v8bf*)(Ar[t] + k + 16 + 8 * lh);
      a[t] = cat8(lo, hi);
      // B fragment 32x16: lane column n=ll, K={0..7,16..23}; lanes 16-31 K=+8
      b[t] = cat8(*(const v8bf*)(Wr[t] + k + 8 * lh), *(const v8bf*)(Wr[t] + k + 16 + 8 * lh));
    }
#pragma unroll
    for (int mt = 0; mt < 4; ++mt)
#pragma unroll
      for (int nt = 0; nt < 4; ++nt)
        acc[mt][nt] = wmma_bf16(a[mt], b[nt], acc[mt][nt]);
  }

#pragma unroll
  for (int nt = 0; nt < 4; ++nt) {
    int col = nBase + nt * 16 + ll;
    float bv = bias[col];
#pragma unroll
    for (int mt = 0; mt < 4; ++mt) {
      int rowBase = mBase + mt * 16 + 8 * lh;  // C layout: VGPR r -> M=r(+8)
#pragma unroll
      for (int r = 0; r < 8; ++r) {
        size_t idx = (size_t)(rowBase + r) * N + col;
        float v = acc[mt][nt][r] + bv;
        if (resid) v += resid[idx];
        C[idx] = v;
      }
    }
  }
}

// ------------------------------------------------- QKV reshape (L,N,E)->(B,L,d)
__global__ __launch_bounds__(256)
void reshape_qkv_kernel(const float* __restrict__ q, const float* __restrict__ k,
                        const float* __restrict__ v, bf16_t* __restrict__ Qb,
                        bf16_t* __restrict__ Kb, bf16_t* __restrict__ Vt,
                        float scale) {
  const int E = 1024, L = 1024, D = 64, H = 16;
  int idx = blockIdx.x * 256 + threadIdx.x;  // over 4096*1024
  int row = idx >> 10;                       // l*N + n
  int col = idx & 1023;                      // h_*64 + di
  int l = row >> 2, n = row & 3;
  int h_ = col >> 6, di = col & 63;
  int b = n * H + h_;
  size_t src = (size_t)row * E + col;
  size_t dst = ((size_t)b * L + l) * D + di;
  Qb[dst] = (bf16_t)(q[src] * scale);
  Kb[dst] = (bf16_t)(k[src]);
  Vt[((size_t)b * D + di) * L + l] = (bf16_t)(v[src]);  // transposed V
}

// ---------------------------------------------------------------- flash attn
// grid (L/64, B=64); block 128 = 4 waves; wave owns 16 query rows.
__global__ __launch_bounds__(128)
void flash_attn_kernel(const bf16_t* __restrict__ Qb, const bf16_t* __restrict__ Kb,
                       const bf16_t* __restrict__ Vt, bf16_t* __restrict__ Ob,
                       int causal) {
  const int L = 1024, D = 64, E = 1024, Nb = 4;
  int b = blockIdx.y;
  int lane = threadIdx.x & 31;
  int wave = threadIdx.x >> 5;
  int lh = lane >> 4, ll = lane & 15;
  int qBase = blockIdx.x * 64 + wave * 16;

  __shared__ bf16_t Pl[4][16 * 32];
  bf16_t* P = Pl[wave];  // per-wave private slab

  const bf16_t* Qrow = Qb + ((size_t)b * L + qBase + ll) * D;
  v16bf aq0 = cat8(*(const v8bf*)(Qrow + 8 * lh), *(const v8bf*)(Qrow + 16 + 8 * lh));
  v16bf aq1 = cat8(*(const v8bf*)(Qrow + 32 + 8 * lh), *(const v8bf*)(Qrow + 48 + 8 * lh));

  v8f o[4] = {};
  float mrow[8], lrow[8];
#pragma unroll
  for (int r = 0; r < 8; ++r) {
    mrow[r] = -3.0e38f;
    lrow[r] = 0.f;
  }

  const int kEnd = causal ? (qBase + 16) : L;
  for (int j0 = 0; j0 < kEnd; j0 += 32) {
    // S = Q*K^T for 32 keys (two 16x16 tiles)
    const bf16_t* K0 = Kb + ((size_t)b * L + j0 + ll) * D;
    const bf16_t* K1 = K0 + 16 * D;
    v8f s0 = {}, s1 = {};
    s0 = wmma_bf16(aq0, cat8(*(const v8bf*)(K0 + 8 * lh), *(const v8bf*)(K0 + 16 + 8 * lh)), s0);
    s0 = wmma_bf16(aq1, cat8(*(const v8bf*)(K0 + 32 + 8 * lh), *(const v8bf*)(K0 + 48 + 8 * lh)), s0);
    s1 = wmma_bf16(aq0, cat8(*(const v8bf*)(K1 + 8 * lh), *(const v8bf*)(K1 + 16 + 8 * lh)), s1);
    s1 = wmma_bf16(aq1, cat8(*(const v8bf*)(K1 + 32 + 8 * lh), *(const v8bf*)(K1 + 48 + 8 * lh)), s1);

    if (causal) {
#pragma unroll
      for (int r = 0; r < 8; ++r) {
        int i = qBase + r + 8 * lh;
        if (j0 + ll > i) s0[r] = -10000.0f;
        if (j0 + 16 + ll > i) s1[r] = -10000.0f;
      }
    }

    float alpha[8];
#pragma unroll
    for (int r = 0; r < 8; ++r) {
      float mv = fmaxf(s0[r], s1[r]);
#pragma unroll
      for (int off = 1; off < 16; off <<= 1) mv = fmaxf(mv, __shfl_xor(mv, off, 16));
      float mnew = fmaxf(mrow[r], mv);
      float p0 = __expf(s0[r] - mnew);
      float p1 = __expf(s1[r] - mnew);
      float ps = p0 + p1;
#pragma unroll
      for (int off = 1; off < 16; off <<= 1) ps += __shfl_xor(ps, off, 16);
      alpha[r] = __expf(mrow[r] - mnew);
      lrow[r] = lrow[r] * alpha[r] + ps;
      mrow[r] = mnew;
      s0[r] = p0;
      s1[r] = p1;
    }
#pragma unroll
    for (int nt = 0; nt < 4; ++nt)
#pragma unroll
      for (int r = 0; r < 8; ++r) o[nt][r] *= alpha[r];

    // C-layout -> A-layout transpose of P via LDS (16x32 bf16, row-major)
#pragma unroll
    for (int r = 0; r < 8; ++r) {
      int prow = r + 8 * lh;
      P[prow * 32 + ll] = (bf16_t)s0[r];
      P[prow * 32 + 16 + ll] = (bf16_t)s1[r];
    }
    asm volatile("s_wait_dscnt 0" ::: "memory");
    const bf16_t* Pr = P + ll * 32;
    v16bf ap = cat8(*(const v8bf*)(Pr + 8 * lh), *(const v8bf*)(Pr + 16 + 8 * lh));

    // O += P * V  (V pre-transposed: row n of Vt is contiguous in key index)
#pragma unroll
    for (int nt = 0; nt < 4; ++nt) {
      const bf16_t* Vr = Vt + ((size_t)b * D + nt * 16 + ll) * L + j0 + 8 * lh;
      o[nt] = wmma_bf16(ap, cat8(*(const v8bf*)Vr, *(const v8bf*)(Vr + 16)), o[nt]);
    }
  }

  int n = b >> 4, h_ = b & 15;  // b = n*H + h_
#pragma unroll
  for (int nt = 0; nt < 4; ++nt)
#pragma unroll
    for (int r = 0; r < 8; ++r) {
      int i = qBase + r + 8 * lh;
      int dcol = nt * 16 + ll;
      Ob[((size_t)i * Nb + n) * E + h_ * 64 + dcol] = (bf16_t)(o[nt][r] / lrow[r]);
    }
}

// ============================================================== host driver
extern "C" void kernel_launch(void* const* d_in, const int* in_sizes, int n_in,
                              void* d_out, int out_size, void* d_ws, size_t ws_size,
                              hipStream_t stream) {
  (void)in_sizes; (void)n_in; (void)out_size; (void)ws_size;
  const int L = 1024, Nb = 4, E = 1024, F = 4096, M = L * Nb;

  const float* x0 = (const float*)d_in[0];
  const float* enc = (const float*)d_in[1];
  const float* wq_s = (const float*)d_in[2];
  const float* bq_s = (const float*)d_in[3];
  const float* wk_s = (const float*)d_in[4];
  const float* bk_s = (const float*)d_in[5];
  const float* wv_s = (const float*)d_in[6];
  const float* bv_s = (const float*)d_in[7];
  const float* wo_s = (const float*)d_in[8];
  const float* bo_s = (const float*)d_in[9];
  const float* wq_c = (const float*)d_in[10];
  const float* bq_c = (const float*)d_in[11];
  const float* wk_c = (const float*)d_in[12];
  const float* bk_c = (const float*)d_in[13];
  const float* wv_c = (const float*)d_in[14];
  const float* bv_c = (const float*)d_in[15];
  const float* wo_c = (const float*)d_in[16];
  const float* bo_c = (const float*)d_in[17];
  const float* ln1_g = (const float*)d_in[18];
  const float* ln1_b = (const float*)d_in[19];
  const float* ln2_g = (const float*)d_in[20];
  const float* ln2_b = (const float*)d_in[21];
  const float* ln3_g = (const float*)d_in[22];
  const float* ln3_b = (const float*)d_in[23];
  const float* fc1_w = (const float*)d_in[24];
  const float* fc1_b = (const float*)d_in[25];
  const float* fc2_w = (const float*)d_in[26];
  const float* fc2_b = (const float*)d_in[27];

  char* ws = (char*)d_ws;
  size_t off = 0;
  auto alloc = [&](size_t bytes) -> void* {
    off = (off + 255) & ~(size_t)255;
    void* p = ws + off;
    off += bytes;
    return p;
  };

  bf16_t* bwq_s = (bf16_t*)alloc((size_t)E * E * 2);
  bf16_t* bwk_s = (bf16_t*)alloc((size_t)E * E * 2);
  bf16_t* bwv_s = (bf16_t*)alloc((size_t)E * E * 2);
  bf16_t* bwo_s = (bf16_t*)alloc((size_t)E * E * 2);
  bf16_t* bwq_c = (bf16_t*)alloc((size_t)E * E * 2);
  bf16_t* bwk_c = (bf16_t*)alloc((size_t)E * E * 2);
  bf16_t* bwv_c = (bf16_t*)alloc((size_t)E * E * 2);
  bf16_t* bwo_c = (bf16_t*)alloc((size_t)E * E * 2);
  bf16_t* bfc1 = (bf16_t*)alloc((size_t)F * E * 2);
  bf16_t* bfc2 = (bf16_t*)alloc((size_t)E * F * 2);
  bf16_t* enc_bf = (bf16_t*)alloc((size_t)M * E * 2);
  bf16_t* hbf = (bf16_t*)alloc((size_t)M * E * 2);
  float* qproj = (float*)alloc((size_t)M * E * 4);
  float* kproj = (float*)alloc((size_t)M * E * 4);
  float* vproj = (float*)alloc((size_t)M * E * 4);
  bf16_t* Qb = (bf16_t*)alloc((size_t)M * E * 2);
  bf16_t* Kb = (bf16_t*)alloc((size_t)M * E * 2);
  bf16_t* Vt = (bf16_t*)alloc((size_t)M * E * 2);
  bf16_t* Ob = (bf16_t*)alloc((size_t)M * E * 2);
  float* x1 = (float*)alloc((size_t)M * E * 4);
  float* x2 = (float*)alloc((size_t)M * E * 4);
  float* ffn_h = (float*)alloc((size_t)M * F * 4);
  bf16_t* gbf = (bf16_t*)alloc((size_t)M * F * 2);

  auto cvt = [&](const float* src, bf16_t* dst, int n) {
    cvt_f32_bf16_kernel<<<n / 1024, 256, 0, stream>>>(src, dst, n);
  };
  cvt(wq_s, bwq_s, E * E);
  cvt(wk_s, bwk_s, E * E);
  cvt(wv_s, bwv_s, E * E);
  cvt(wo_s, bwo_s, E * E);
  cvt(wq_c, bwq_c, E * E);
  cvt(wk_c, bwk_c, E * E);
  cvt(wv_c, bwv_c, E * E);
  cvt(wo_c, bwo_c, E * E);
  cvt(fc1_w, bfc1, F * E);
  cvt(fc2_w, bfc2, E * F);
  cvt(enc, enc_bf, M * E);

  dim3 gproj(E / 256, M / 128);  // N=1024 GEMMs
  dim3 gfc1(F / 256, M / 128);   // N=4096 GEMM
  dim3 gattn(L / 64, Nb * 16);
  const float scale = 0.125f;  // 64^-0.5

  // ---- self attention (causal) ----
  layernorm_bf16_kernel<<<M, 256, 0, stream>>>(x0, ln1_g, ln1_b, hbf);
  gemm_bf16_wmma_kernel<<<gproj, 256, 0, stream>>>(hbf, bwq_s, bq_s, nullptr, qproj, M, E, E);
  gemm_bf16_wmma_kernel<<<gproj, 256, 0, stream>>>(hbf, bwk_s, bk_s, nullptr, kproj, M, E, E);
  gemm_bf16_wmma_kernel<<<gproj, 256, 0, stream>>>(hbf, bwv_s, bv_s, nullptr, vproj, M, E, E);
  reshape_qkv_kernel<<<(M * E) / 256, 256, 0, stream>>>(qproj, kproj, vproj, Qb, Kb, Vt, scale);
  flash_attn_kernel<<<gattn, 128, 0, stream>>>(Qb, Kb, Vt, Ob, 1);
  gemm_bf16_wmma_kernel<<<gproj, 256, 0, stream>>>(Ob, bwo_s, bo_s, x0, x1, M, E, E);

  // ---- cross attention ----
  layernorm_bf16_kernel<<<M, 256, 0, stream>>>(x1, ln2_g, ln2_b, hbf);
  gemm_bf16_wmma_kernel<<<gproj, 256, 0, stream>>>(hbf, bwq_c, bq_c, nullptr, qproj, M, E, E);
  gemm_bf16_wmma_kernel<<<gproj, 256, 0, stream>>>(enc_bf, bwk_c, bk_c, nullptr, kproj, M, E, E);
  gemm_bf16_wmma_kernel<<<gproj, 256, 0, stream>>>(enc_bf, bwv_c, bv_c, nullptr, vproj, M, E, E);
  reshape_qkv_kernel<<<(M * E) / 256, 256, 0, stream>>>(qproj, kproj, vproj, Qb, Kb, Vt, scale);
  flash_attn_kernel<<<gattn, 128, 0, stream>>>(Qb, Kb, Vt, Ob, 0);
  gemm_bf16_wmma_kernel<<<gproj, 256, 0, stream>>>(Ob, bwo_c, bo_c, x1, x2, M, E, E);

  // ---- FFN ----
  layernorm_bf16_kernel<<<M, 256, 0, stream>>>(x2, ln3_g, ln3_b, hbf);
  gemm_bf16_wmma_kernel<<<gfc1, 256, 0, stream>>>(hbf, bfc1, fc1_b, nullptr, ffn_h, M, F, E);
  gelu_f32_bf16_kernel<<<(M * F) / 256, 256, 0, stream>>>(ffn_h, gbf, M * F);
  gemm_bf16_wmma_kernel<<<gproj, 256, 0, stream>>>(gbf, bfc2, fc2_b, x2, (float*)d_out, M, E, F);
}
